// GATConv_60962765799926
// MI455X (gfx1250) — matrix operand order repaired and checked by reading, not compiled, not run on hardware
//
#include <hip/hip_runtime.h>
#include <hip/hip_bf16.h>
#include <stdint.h>

// ---------------------------------------------------------------------------
// GATConv forward for MI455X (gfx1250, wave32).
//   feat:[N,128] f32, W:[256,128] f32, attn_l/attn_r:[8,32] f32,
//   src/dst:[E] i32  ->  out:[N,8,32] f32
// FC projection runs on v_wmma_f32_16x16x32_bf16 (W pre-converted to bf16
// once; A converted on the fly since feat is streamed). Edge softmax /
// aggregation is L2-resident atomics (192MB L2 holds the working set).
// ---------------------------------------------------------------------------

typedef __attribute__((ext_vector_type(16))) __bf16 v16bf;
typedef __attribute__((ext_vector_type(8)))  float  v8f;

#define HD 256   // HEADS * OUT_FEATS
#define KF 128   // IN_FEATS
#define NHEAD 8
#define DHEAD 32

// ---- monotone float<->uint encoding for atomic float max ------------------
__device__ __forceinline__ unsigned enc_f(float f) {
  unsigned u = __float_as_uint(f);
  return (u & 0x80000000u) ? ~u : (u | 0x80000000u);
}
__device__ __forceinline__ float dec_f(unsigned u) {
  return __uint_as_float((u & 0x80000000u) ? (u ^ 0x80000000u) : ~u);
}
#define ENC_NEG_INF 0x007FFFFFu   // enc_f(-inf)

// ---------------------------------------------------------------------------
// 0) init: zero d_out, zero z, set segment-max accumulators to enc(-inf),
//    and convert W (32K elements) to bf16 exactly once per launch.
// ---------------------------------------------------------------------------
__global__ __launch_bounds__(256) void gat_init(float* __restrict__ out,
                                                float* __restrict__ z,
                                                unsigned* __restrict__ mEnc,
                                                const float* __restrict__ W,
                                                __bf16* __restrict__ Wb,
                                                long long outN, int NH, int WN) {
  long long i = (long long)blockIdx.x * 256 + threadIdx.x;
  if (i < outN) out[i] = 0.0f;
  if (i < NH) { z[i] = 0.0f; mEnc[i] = ENC_NEG_INF; }
  if (i < WN) Wb[i] = (__bf16)W[i];
}

// ---------------------------------------------------------------------------
// 1) FC projection: feat_hd = feat @ W^T via WMMA bf16, fp32 accumulate.
//    block = 256 (8 waves); each wave: one 16-row tile x two 16-col tiles.
//    B fragments load pre-converted bf16 (32B v16bf loads, no cvt in loop).
// ---------------------------------------------------------------------------
__global__ __launch_bounds__(256) void gat_fc_wmma(const float* __restrict__ feat,
                                                   const __bf16* __restrict__ Wb,
                                                   float* __restrict__ feat_hd,
                                                   int Nn) {
  const int wave = threadIdx.x >> 5;       // 0..7
  const int lane = threadIdx.x & 31;
  const int row0 = blockIdx.x * 16;        // node tile base
  const int mrow = lane & 15;
  const int lhi  = lane >> 4;              // 0 or 1 (lane half)

  // A-source row (clamped; N is a multiple of 16 in the reference)
  int rm = row0 + mrow; if (rm >= Nn) rm = Nn - 1;
  const float*  __restrict__ arow  = feat + (size_t)rm * KF;
  const __bf16* __restrict__ brow0 = Wb + (size_t)(wave * 16 + mrow) * KF;
  const __bf16* __restrict__ brow1 = Wb + (size_t)((wave + 8) * 16 + mrow) * KF;

  v8f acc0 = {}; v8f acc1 = {};

  #pragma unroll
  for (int ks = 0; ks < 4; ++ks) {
    // A fragment: K = {kbA..kbA+15}, kbA = 32ks + 16*lhi
    const int kbA = ks * 32 + lhi * 16;
    // B fragment: 16 contiguous K starting at 32ks + 16*lhi (32B aligned)
    const int kbB = ks * 32 + lhi * 16;
    v16bf af;
    #pragma unroll
    for (int j = 0; j < 8; ++j) {
      af[j]     = (__bf16)arow[kbA + j];
      af[j + 8] = (__bf16)arow[kbA + 8 + j];
    }
    const v16bf b0 = *(const v16bf*)(brow0 + kbB);
    const v16bf b1 = *(const v16bf*)(brow1 + kbB);
    acc0 = __builtin_amdgcn_wmma_f32_16x16x32_bf16(false, af, false, b0,
                                                   (short)0, acc0, false, false);
    acc1 = __builtin_amdgcn_wmma_f32_16x16x32_bf16(false, af, false, b1,
                                                   (short)0, acc1, false, false);
  }

  // D layout: lane holds column n = lane&15; rows 8*lhi + r (r = VGPR index)
  const int ncol0 = wave * 16 + (lane & 15);
  const int ncol1 = (wave + 8) * 16 + (lane & 15);
  const int rbase = lhi * 8;
  #pragma unroll
  for (int r = 0; r < 8; ++r) {
    int m = row0 + rbase + r;
    if (m < Nn) {
      feat_hd[(size_t)m * HD + ncol0] = acc0[r];
      feat_hd[(size_t)m * HD + ncol1] = acc1[r];
    }
  }
}

// ---------------------------------------------------------------------------
// 2) per-(node,head) attention logits: el = <feat_hd, attn_l>, er likewise
// ---------------------------------------------------------------------------
__global__ __launch_bounds__(256) void gat_logits(const float* __restrict__ feat_hd,
                                                  const float* __restrict__ al,
                                                  const float* __restrict__ ar,
                                                  float* __restrict__ el,
                                                  float* __restrict__ er, int NH) {
  int i = blockIdx.x * 256 + threadIdx.x;
  if (i >= NH) return;
  int n = i >> 3, h = i & 7;
  const float* f = feat_hd + (size_t)n * HD + h * DHEAD;
  const float* A = al + h * DHEAD;
  const float* B = ar + h * DHEAD;
  float sl = 0.f, sr = 0.f;
  #pragma unroll
  for (int d = 0; d < DHEAD; ++d) { float v = f[d]; sl += v * A[d]; sr += v * B[d]; }
  el[i] = sl; er[i] = sr;
}

// ---------------------------------------------------------------------------
// 3) edge scores: leaky-relu(el[src]+er[dst]); segment max into mEnc[dst]
// ---------------------------------------------------------------------------
__global__ __launch_bounds__(256) void gat_edge_max(const float* __restrict__ el,
                                                    const float* __restrict__ er,
                                                    const int* __restrict__ src,
                                                    const int* __restrict__ dst,
                                                    float* __restrict__ esc,
                                                    unsigned* __restrict__ mEnc, int EH) {
  int i = blockIdx.x * 256 + threadIdx.x;
  if (i >= EH) return;
  int e = i >> 3, h = i & 7;
  int d0 = dst[e];
  float s = el[(size_t)src[e] * NHEAD + h] + er[(size_t)d0 * NHEAD + h];
  s = (s > 0.f) ? s : 0.2f * s;
  esc[i] = s;
  atomicMax(mEnc + (size_t)d0 * NHEAD + h, enc_f(s));
}

// ---------------------------------------------------------------------------
// 4) exp(e - max) and segment sum into z[dst]
// ---------------------------------------------------------------------------
__global__ __launch_bounds__(256) void gat_edge_exp(const int* __restrict__ dst,
                                                    const unsigned* __restrict__ mEnc,
                                                    float* __restrict__ esc,
                                                    float* __restrict__ z, int EH) {
  int i = blockIdx.x * 256 + threadIdx.x;
  if (i >= EH) return;
  int e = i >> 3, h = i & 7;
  size_t di = (size_t)dst[e] * NHEAD + h;
  float a = __expf(esc[i] - dec_f(mEnc[di]));
  esc[i] = a;
  atomicAdd(z + di, a);
}

// ---------------------------------------------------------------------------
// 5) weighted scatter-aggregation: out[dst] += feat_hd[src] * (a/z[dst])
//    one block per edge, 256 threads = (head, dim); weights staged in LDS.
// ---------------------------------------------------------------------------
__global__ __launch_bounds__(256) void gat_aggregate(const float* __restrict__ feat_hd,
                                                     const float* __restrict__ esc,
                                                     const float* __restrict__ z,
                                                     const int* __restrict__ src,
                                                     const int* __restrict__ dst,
                                                     float* __restrict__ out) {
  const int e = blockIdx.x;
  const int t = threadIdx.x;
  __shared__ float wsh[NHEAD];
  const int s  = src[e];
  const int d0 = dst[e];
  if (t < NHEAD)
    wsh[t] = esc[(size_t)e * NHEAD + t] / z[(size_t)d0 * NHEAD + t];
  __syncthreads();
  const float v = feat_hd[(size_t)s * HD + t] * wsh[t >> 5];
  atomicAdd(out + (size_t)d0 * HD + t, v);
}

// ---------------------------------------------------------------------------
extern "C" void kernel_launch(void* const* d_in, const int* in_sizes, int n_in,
                              void* d_out, int out_size, void* d_ws, size_t ws_size,
                              hipStream_t stream) {
  const float* feat = (const float*)d_in[0];
  const float* W    = (const float*)d_in[1];
  const float* al   = (const float*)d_in[2];
  const float* ar   = (const float*)d_in[3];
  const int*   src  = (const int*)d_in[4];
  const int*   dst  = (const int*)d_in[5];
  float* out = (float*)d_out;

  const int Nn = in_sizes[0] / KF;     // 50000
  const int E  = in_sizes[4];          // 800000
  const int WN = in_sizes[1];          // 256*128
  const int NH = Nn * NHEAD;
  const int EH = E * NHEAD;
  const long long outN = (long long)Nn * HD;

  // workspace layout (floats unless noted)
  float*    feat_hd = (float*)d_ws;                 // Nn*256
  float*    el      = feat_hd + (size_t)Nn * HD;    // Nn*8
  float*    er      = el + (size_t)NH;              // Nn*8
  float*    z       = er + (size_t)NH;              // Nn*8
  unsigned* mEnc    = (unsigned*)(z + (size_t)NH);  // Nn*8
  float*    esc     = (float*)(mEnc + (size_t)NH);  // E*8
  // bf16 W copy, rounded up to 32B alignment for v16bf loads
  uintptr_t wbaddr  = (uintptr_t)(esc + (size_t)EH);
  __bf16*   Wb      = (__bf16*)((wbaddr + 31) & ~(uintptr_t)31);

  dim3 blk(256);

  // 0) init accumulators + one-time bf16 conversion of W (idempotent launch)
  gat_init<<<dim3((unsigned)((outN + 255) / 256)), blk, 0, stream>>>(
      out, z, mEnc, W, Wb, outN, NH, WN);

  // 1) WMMA FC projection
  gat_fc_wmma<<<dim3((unsigned)((Nn + 15) / 16)), blk, 0, stream>>>(feat, Wb, feat_hd, Nn);

  // 2) attention logits
  gat_logits<<<dim3((unsigned)((NH + 255) / 256)), blk, 0, stream>>>(feat_hd, al, ar, el, er, NH);

  // 3) edge scores + segment max
  gat_edge_max<<<dim3((unsigned)((EH + 255) / 256)), blk, 0, stream>>>(el, er, src, dst, esc, mEnc, EH);

  // 4) exp + segment sum
  gat_edge_exp<<<dim3((unsigned)((EH + 255) / 256)), blk, 0, stream>>>(dst, mEnc, esc, z, EH);

  // 5) weighted scatter-aggregation
  gat_aggregate<<<dim3((unsigned)E), blk, 0, stream>>>(feat_hd, esc, z, src, dst, out);
}
